// MLAAttention_27376121544933
// MI455X (gfx1250) — matrix operand order repaired and checked by reading, not compiled, not run on hardware
//
#include <hip/hip_runtime.h>
#include <cstdint>

// ---- problem constants --------------------------------------------------
#define NH     16      // heads
#define EMB    2048
#define QRANK  1536
#define KVRANK 512
#define ROPED  64
#define QKD    128     // nope(64)+rope(64)
#define VDIM   128
#define KVBD   192     // nope(64)+v(128)

typedef unsigned short u16;
typedef __attribute__((ext_vector_type(16))) __bf16 bf16x16;
typedef __attribute__((ext_vector_type(8)))  float  f32x8;

union B16V { bf16x16 v; uint4 q[2]; };
union U4U  { uint4 q; u16 u[8]; };

__device__ __forceinline__ u16 f2bf(float f) {
  union { float f; uint32_t u; } c; c.f = f;
  uint32_t r = c.u + 0x7FFFu + ((c.u >> 16) & 1u);   // RNE
  return (u16)(r >> 16);
}
__device__ __forceinline__ float bf2f(u16 h) {
  union { float f; uint32_t u; } c; c.u = ((uint32_t)h) << 16; return c.f;
}
__device__ __forceinline__ f32x8 zero8() { f32x8 z; for (int e = 0; e < 8; ++e) z[e] = 0.f; return z; }

#define WMMA_BF16(A_, B_, C_) \
  __builtin_amdgcn_wmma_f32_16x16x32_bf16(false, (A_), false, (B_), (short)0, (C_), false, false)

// ---- fp32 -> bf16 conversion -------------------------------------------
__global__ void cvt_f32_bf16(const float* __restrict__ in, u16* __restrict__ out, size_t n) {
  size_t i = (size_t)blockIdx.x * blockDim.x + threadIdx.x;
  size_t st = (size_t)gridDim.x * blockDim.x;
  for (; i < n; i += st) out[i] = f2bf(in[i]);
}

// ---- generic bf16 GEMM:  C[M,N] = A[M,K] * W[N,K]^T ---------------------
// Block tile 128x64, BK=32, 128 threads (4 waves in a 2x2 grid).
// Wave tile 64x32: 4 M-tiles x 2 N-tiles = 8 accumulators.
// Register-staged global->LDS double buffering: loads for k+1 overlap WMMAs of k.
template<bool BF16OUT>
__global__ __launch_bounds__(128, 1)
void gemm_bf16_wmma(const u16* __restrict__ A, int lda,
                    const u16* __restrict__ W, int ldw,
                    void* __restrict__ C, int ldc,
                    int M, int N, int K) {
  __shared__ __align__(16) u16 As[2][128][40];   // 40*2B rows -> 16B aligned, bank-clean
  __shared__ __align__(16) u16 Bs[2][64][40];

  const int tid  = threadIdx.x;
  const int wave = tid >> 5;
  const int lane = tid & 31;
  const int g    = lane >> 4;
  const int ln   = lane & 15;
  const int wm   = (wave >> 1) * 64;   // wave M offset in block tile
  const int wn   = (wave & 1) * 32;    // wave N offset in block tile
  const int mblk = blockIdx.y * 128;
  const int nblk = blockIdx.x * 64;

  f32x8 acc[4][2];
  for (int mt = 0; mt < 4; ++mt)
    for (int nt = 0; nt < 2; ++nt) acc[mt][nt] = zero8();

  const int alr = tid >> 1;          // loader row 0..63
  const int alc = (tid & 1) * 16;    // 16-element half-row

  uint4 a0, a1, a2, a3, b0, b1;      // staging registers for next tile
  auto gload = [&](int k0) {
    const u16* ag0 = A + (size_t)(mblk + alr) * lda + k0 + alc;
    a0 = ((const uint4*)ag0)[0]; a1 = ((const uint4*)ag0)[1];
    const u16* ag1 = A + (size_t)(mblk + 64 + alr) * lda + k0 + alc;
    a2 = ((const uint4*)ag1)[0]; a3 = ((const uint4*)ag1)[1];
    const u16* wg = W + (size_t)(nblk + alr) * ldw + k0 + alc;
    b0 = ((const uint4*)wg)[0]; b1 = ((const uint4*)wg)[1];
  };
  auto sstore = [&](int buf) {
    ((uint4*)&As[buf][alr][alc])[0]      = a0;
    ((uint4*)&As[buf][alr][alc])[1]      = a1;
    ((uint4*)&As[buf][64 + alr][alc])[0] = a2;
    ((uint4*)&As[buf][64 + alr][alc])[1] = a3;
    ((uint4*)&Bs[buf][alr][alc])[0]      = b0;
    ((uint4*)&Bs[buf][alr][alc])[1]      = b1;
  };

  gload(0); sstore(0);
  __syncthreads();

  int buf = 0;
  for (int k0 = 0; k0 < K; k0 += 32) {
    const bool more = (k0 + 32 < K);
    if (more) gload(k0 + 32);        // global loads hidden under the 8 WMMAs below

    B16V bv[2];                      // B 32x16: lane holds K=16g..16g+15, col ln
    for (int nt = 0; nt < 2; ++nt) {
      bv[nt].q[0] = ((const uint4*)&Bs[buf][wn + nt * 16 + ln][16 * g])[0];
      bv[nt].q[1] = ((const uint4*)&Bs[buf][wn + nt * 16 + ln][16 * g])[1];
    }
    for (int mt = 0; mt < 4; ++mt) {
      B16V av;                       // A 16x32: lane holds M=ln, K=16g..16g+15
      av.q[0] = *(const uint4*)&As[buf][wm + mt * 16 + ln][16 * g];
      av.q[1] = *(const uint4*)&As[buf][wm + mt * 16 + ln][16 * g + 8];
      acc[mt][0] = WMMA_BF16(av.v, bv[0].v, acc[mt][0]);
      acc[mt][1] = WMMA_BF16(av.v, bv[1].v, acc[mt][1]);
    }
    if (more) sstore(buf ^ 1);
    __syncthreads();
    buf ^= 1;
  }

  for (int mt = 0; mt < 4; ++mt)
    for (int nt = 0; nt < 2; ++nt) {
      const int n = nblk + wn + nt * 16 + ln;
      for (int r = 0; r < 8; ++r) {                // D element r -> row r+8g, col ln
        int m = mblk + wm + mt * 16 + r + 8 * g;
        float v = acc[mt][nt][r];
        if (BF16OUT) ((u16*)C)[(size_t)m * ldc + n] = f2bf(v);
        else         ((float*)C)[(size_t)m * ldc + n] = v;
      }
    }
}

// ---- RoPE on q_rot; repack q (B,S,H,128)bf16 -> query (B,H,S,128)bf16 ---
__global__ void rope_q_kernel(const u16* __restrict__ qfull, u16* __restrict__ query, int B, int S) {
  size_t idx = (size_t)blockIdx.x * blockDim.x + threadIdx.x;
  size_t total = (size_t)B * S * NH * 64;
  if (idx >= total) return;
  int j = idx & 63; size_t t = idx >> 6;
  int h = (int)(t % NH); size_t bs = t / NH;
  int s = (int)(bs % S); int b = (int)(bs / S);
  const u16* src = qfull + (bs * NH + h) * QKD;
  u16* dst = query + (((size_t)b * NH + h) * S + s) * QKD;
  if (j < 32) {                       // pass-through half
    dst[2 * j] = src[2 * j];
    dst[2 * j + 1] = src[2 * j + 1];
  } else {                            // rope half (interleaved pairs)
    int i = j - 32;
    float theta = __powf(10000.f, -(float)i / 32.f);
    float sn, cs; __sincosf((float)s * theta, &sn, &cs);
    float x0 = bf2f(src[64 + 2 * i]), x1 = bf2f(src[64 + 2 * i + 1]);
    dst[64 + 2 * i]     = f2bf(x0 * cs - x1 * sn);
    dst[64 + 2 * i + 1] = f2bf(x1 * cs + x0 * sn);
  }
}

// ---- RoPE on k_rot (last 64 cols of ckv) -> krot (B*S,64) ---------------
__global__ void rope_k_kernel(const u16* __restrict__ ckv, u16* __restrict__ krot, int B, int S) {
  size_t idx = (size_t)blockIdx.x * blockDim.x + threadIdx.x;
  size_t total = (size_t)B * S * 32;
  if (idx >= total) return;
  int i = idx & 31; size_t bs = idx >> 5;
  int s = (int)(bs % S);
  const u16* src = ckv + bs * (KVRANK + ROPED) + KVRANK;
  float theta = __powf(10000.f, -(float)i / 32.f);
  float sn, cs; __sincosf((float)s * theta, &sn, &cs);
  float x0 = bf2f(src[2 * i]), x1 = bf2f(src[2 * i + 1]);
  krot[bs * 64 + 2 * i]     = f2bf(x0 * cs - x1 * sn);
  krot[bs * 64 + 2 * i + 1] = f2bf(x1 * cs + x0 * sn);
}

// ---- split kvb (B,S,H,192) -> key (B,H,S,128) (k_pass|krot) and v -------
__global__ void split_kv_kernel(const u16* __restrict__ kvb, const u16* __restrict__ krot,
                                u16* __restrict__ key, u16* __restrict__ vmat, int B, int S) {
  size_t idx = (size_t)blockIdx.x * blockDim.x + threadIdx.x;
  size_t total = (size_t)B * S * NH * 256;
  if (idx >= total) return;
  int j = idx & 255; size_t t = idx >> 8;
  int h = (int)(t % NH); size_t bs = t / NH;
  int s = (int)(bs % S); int b = (int)(bs / S);
  const u16* kv = kvb + bs * (NH * KVBD) + (size_t)h * KVBD;
  size_t rowbh = (((size_t)b * NH + h) * S + s) * 128;
  if (j < 64)        key[rowbh + j] = kv[j];                    // k_pass
  else if (j < 128)  key[rowbh + j] = krot[bs * 64 + (j - 64)]; // broadcast rope
  else               vmat[rowbh + (j - 128)] = kv[64 + (j - 128)];
}

// ---- flash attention: one block per (b,h,64 q rows), 4 waves ------------
__global__ __launch_bounds__(128, 1)
void flash_attn_wmma(const u16* __restrict__ query, const u16* __restrict__ key,
                     const u16* __restrict__ vmat, u16* __restrict__ attn_out,
                     int B, int S) {
  __shared__ __align__(16) u16 Qs[64][136];
  __shared__ __align__(16) u16 Ks[64][136];
  __shared__ __align__(16) u16 Vts[128][72];   // V transposed: [d][key]
  __shared__ __align__(16) u16 Ps[4][16][72];  // per-wave P tile

  const int tid  = threadIdx.x;
  const int wave = tid >> 5;
  const int lane = tid & 31;
  const int g    = lane >> 4;
  const int ln   = lane & 15;
  const int qt = blockIdx.x, h = blockIdx.y, b = blockIdx.z;
  const size_t basebh = ((size_t)b * NH + h) * S;
  const int qbase = qt * 64;

  // Q tile -> LDS
  for (int idx = tid; idx < 64 * 16; idx += 128) {
    int row = idx >> 4, col = (idx & 15) * 8;
    *(uint4*)&Qs[row][col] = *(const uint4*)(query + (basebh + qbase + row) * 128 + col);
  }

  f32x8 oacc[8];
  for (int dt = 0; dt < 8; ++dt) oacc[dt] = zero8();
  float mstat[8], lstat[8];
  for (int r = 0; r < 8; ++r) { mstat[r] = -3.0e38f; lstat[r] = 0.f; }

  for (int kt = 0; kt < S / 64; ++kt) {
    __syncthreads();
    for (int idx = tid; idx < 64 * 16; idx += 128) {     // K tile
      int row = idx >> 4, col = (idx & 15) * 8;
      *(uint4*)&Ks[row][col] = *(const uint4*)(key + (basebh + kt * 64 + row) * 128 + col);
    }
    for (int idx = tid; idx < 64 * 16; idx += 128) {     // V tile, transposed store
      int k = idx >> 4, dc = (idx & 15) * 8;
      U4U t; t.q = *(const uint4*)(vmat + (basebh + kt * 64 + k) * 128 + dc);
      for (int e = 0; e < 8; ++e) Vts[dc + e][k] = t.u[e];
    }
    __syncthreads();

    // scores: S = Q(16x128) * K^T(128x64), per wave
    f32x8 sacc[4];
    for (int nt = 0; nt < 4; ++nt) sacc[nt] = zero8();
    for (int dc = 0; dc < 4; ++dc) {
      B16V av;
      av.q[0] = *(const uint4*)&Qs[wave * 16 + ln][dc * 32 + 16 * g];
      av.q[1] = *(const uint4*)&Qs[wave * 16 + ln][dc * 32 + 16 * g + 8];
      for (int nt = 0; nt < 4; ++nt) {
        B16V bv;
        bv.q[0] = *(const uint4*)&Ks[nt * 16 + ln][dc * 32 + 16 * g];
        bv.q[1] = *(const uint4*)&Ks[nt * 16 + ln][dc * 32 + 16 * g + 8];
        sacc[nt] = WMMA_BF16(av.v, bv.v, sacc[nt]);
      }
    }

    // online softmax (row stats: element r <-> q row r+8g; reduce across 16-lane half)
    for (int r = 0; r < 8; ++r) {
      float mx = sacc[0][r];
      for (int nt = 1; nt < 4; ++nt) mx = fmaxf(mx, sacc[nt][r]);
      for (int off = 1; off < 16; off <<= 1) mx = fmaxf(mx, __shfl_xor(mx, off, 32));
      float mnew  = fmaxf(mstat[r], mx);
      float alpha = __expf(mstat[r] - mnew);
      float rs = 0.f;
      for (int nt = 0; nt < 4; ++nt) {
        float p = __expf(sacc[nt][r] - mnew);
        rs += p;
        Ps[wave][r + 8 * g][nt * 16 + ln] = f2bf(p);   // D-layout -> A-layout via LDS
      }
      for (int off = 1; off < 16; off <<= 1) rs += __shfl_xor(rs, off, 32);
      lstat[r] = lstat[r] * alpha + rs;
      mstat[r] = mnew;
      for (int dt = 0; dt < 8; ++dt) oacc[dt][r] *= alpha;
    }
    asm volatile("s_wait_dscnt 0" ::: "memory");   // P stores visible before A-layout reads

    // O += P(16x64) * V(64x128)
    for (int kc = 0; kc < 2; ++kc) {
      B16V av;
      av.q[0] = *(const uint4*)&Ps[wave][ln][kc * 32 + 16 * g];
      av.q[1] = *(const uint4*)&Ps[wave][ln][kc * 32 + 16 * g + 8];
      for (int dt = 0; dt < 8; ++dt) {
        B16V bv;
        bv.q[0] = *(const uint4*)&Vts[dt * 16 + ln][kc * 32 + 16 * g];
        bv.q[1] = *(const uint4*)&Vts[dt * 16 + ln][kc * 32 + 16 * g + 8];
        oacc[dt] = WMMA_BF16(av.v, bv.v, oacc[dt]);
      }
    }
  }

  // epilogue: normalize, write attn_out as (B,S,H*128) bf16
  for (int dt = 0; dt < 8; ++dt)
    for (int r = 0; r < 8; ++r) {
      int qrow = qbase + wave * 16 + r + 8 * g;
      int d = dt * 16 + ln;
      attn_out[((size_t)b * S + qrow) * (NH * 128) + h * 128 + d] = f2bf(oacc[dt][r] / lstat[r]);
    }
}

// ---- driver --------------------------------------------------------------
extern "C" void kernel_launch(void* const* d_in, const int* in_sizes, int n_in,
                              void* d_out, int out_size, void* d_ws, size_t ws_size,
                              hipStream_t stream) {
  (void)in_sizes; (void)n_in; (void)out_size; (void)ws_size;
  const float* x    = (const float*)d_in[0];
  const float* qaw  = (const float*)d_in[1];
  const float* qbw  = (const float*)d_in[2];
  const float* kvaw = (const float*)d_in[3];
  const float* kvbw = (const float*)d_in[4];
  const float* ow   = (const float*)d_in[5];

  const int B = 2, S = 2048, M = B * S;

  char* ws = (char*)d_ws;
  size_t off = 0;
  auto take = [&](size_t elems) -> u16* {
    u16* p = (u16*)(ws + off);
    off = (off + elems * 2 + 255) & ~(size_t)255;
    return p;
  };
  u16* x_bf    = take((size_t)M * EMB);
  u16* qaw_bf  = take((size_t)QRANK * EMB);
  u16* qbw_bf  = take((size_t)(NH * QKD) * QRANK);
  u16* kvaw_bf = take((size_t)(KVRANK + ROPED) * EMB);
  u16* kvbw_bf = take((size_t)(NH * KVBD) * KVRANK);
  u16* ow_bf   = take((size_t)EMB * (NH * VDIM));
  u16* cq_bf   = take((size_t)M * QRANK);
  u16* qfull   = take((size_t)M * (NH * QKD));
  u16* queryb  = take((size_t)M * (NH * QKD));
  u16* ckv_bf  = take((size_t)M * (KVRANK + ROPED));
  u16* krot_bf = take((size_t)M * ROPED);
  u16* kvb_bf  = take((size_t)M * (NH * KVBD));
  u16* key_bf  = take((size_t)M * (NH * 128));
  u16* v_bf    = take((size_t)M * (NH * 128));
  u16* ao_bf   = take((size_t)M * (NH * VDIM));

  // 1) fp32 -> bf16 conversions
  cvt_f32_bf16<<<4096, 256, 0, stream>>>(x,    x_bf,    (size_t)M * EMB);
  cvt_f32_bf16<<<4096, 256, 0, stream>>>(qaw,  qaw_bf,  (size_t)QRANK * EMB);
  cvt_f32_bf16<<<4096, 256, 0, stream>>>(qbw,  qbw_bf,  (size_t)(NH * QKD) * QRANK);
  cvt_f32_bf16<<<2048, 256, 0, stream>>>(kvaw, kvaw_bf, (size_t)(KVRANK + ROPED) * EMB);
  cvt_f32_bf16<<<2048, 256, 0, stream>>>(kvbw, kvbw_bf, (size_t)(NH * KVBD) * KVRANK);
  cvt_f32_bf16<<<4096, 256, 0, stream>>>(ow,   ow_bf,   (size_t)EMB * (NH * VDIM));

  // 2) cq = x @ q_a_w^T    (M x 1536, K=2048)
  gemm_bf16_wmma<true><<<dim3(QRANK / 64, M / 128), 128, 0, stream>>>(
      x_bf, EMB, qaw_bf, EMB, cq_bf, QRANK, M, QRANK, EMB);
  // 3) qfull = cq @ q_b_w^T (M x 2048, K=1536)
  gemm_bf16_wmma<true><<<dim3((NH * QKD) / 64, M / 128), 128, 0, stream>>>(
      cq_bf, QRANK, qbw_bf, QRANK, qfull, NH * QKD, M, NH * QKD, QRANK);
  // 4) ckv = x @ kv_a_w^T  (M x 576, K=2048)
  gemm_bf16_wmma<true><<<dim3((KVRANK + ROPED) / 64, M / 128), 128, 0, stream>>>(
      x_bf, EMB, kvaw_bf, EMB, ckv_bf, KVRANK + ROPED, M, KVRANK + ROPED, EMB);
  // 5) kvb = ckv[:, :512] @ kv_b_w^T (M x 3072, K=512); lda skips rope cols
  gemm_bf16_wmma<true><<<dim3((NH * KVBD) / 64, M / 128), 128, 0, stream>>>(
      ckv_bf, KVRANK + ROPED, kvbw_bf, KVRANK, kvb_bf, NH * KVBD, M, NH * KVBD, KVRANK);

  // 6) RoPE + layout shuffles
  {
    size_t tq = (size_t)M * NH * 64;
    rope_q_kernel<<<(unsigned)((tq + 255) / 256), 256, 0, stream>>>(qfull, queryb, B, S);
    size_t tk = (size_t)M * 32;
    rope_k_kernel<<<(unsigned)((tk + 255) / 256), 256, 0, stream>>>(ckv_bf, krot_bf, B, S);
    size_t ts = (size_t)M * NH * 256;
    split_kv_kernel<<<(unsigned)((ts + 255) / 256), 256, 0, stream>>>(kvb_bf, krot_bf, key_bf, v_bf, B, S);
  }

  // 7) flash attention
  flash_attn_wmma<<<dim3(S / 64, NH, B), 128, 0, stream>>>(queryb, key_bf, v_bf, ao_bf, B, S);

  // 8) out = attn_out @ o_w^T  (M x 2048, K=2048), f32 to d_out
  gemm_bf16_wmma<false><<<dim3(EMB / 64, M / 128), 128, 0, stream>>>(
      ao_bf, NH * VDIM, ow_bf, EMB, d_out, EMB, M, EMB, NH * VDIM);
}
